// MambaBlock_52080773432047
// MI455X (gfx1250) — compile-verified, matrix operand order checked by reading, not run on hardware
//
#include <hip/hip_runtime.h>

#define D_MODEL   1024
#define N_HEADS   32
#define D_INNER   2048
#define HEADDIM   64
#define D_STATE   128
#define CONV_DIM  2304            // D_INNER + 2*D_STATE
#define D_IN_PROJ 4384            // 2*D_INNER + 2*D_STATE + N_HEADS
#define D_FF      4096
#define BATCH     2
#define SEQLEN    2048
#define NROWS     (BATCH*SEQLEN)  // 4096

typedef __attribute__((ext_vector_type(16))) __bf16        v16bf;
typedef __attribute__((ext_vector_type(8)))  float         v8f;
typedef __attribute__((ext_vector_type(4)))  unsigned int  v4u;

union Frag { v4u u[2]; v16bf v; };

__device__ __forceinline__ unsigned short f2bf(float f) {
  unsigned int u = __float_as_uint(f);
  u += 0x7FFFu + ((u >> 16) & 1u);          // round-to-nearest-even
  return (unsigned short)(u >> 16);
}

// CDNA5 async global->LDS copy (ASYNCcnt-tracked), 16 bytes per lane.
__device__ __forceinline__ void async_copy16(unsigned lds_addr,
                                             const unsigned short* gaddr) {
  asm volatile("global_load_async_to_lds_b128 %0, %1, off"
               :: "v"(lds_addr), "v"(gaddr) : "memory");
}
__device__ __forceinline__ void wait_async0() {
  asm volatile("s_wait_asynccnt 0" ::: "memory");
}

__device__ __forceinline__ float block_sum256(float v) {
  #pragma unroll
  for (int o = 16; o > 0; o >>= 1) v += __shfl_xor(v, o, 32);
  __shared__ float smem[8];
  const int lane = threadIdx.x & 31, w = threadIdx.x >> 5;
  if (lane == 0) smem[w] = v;
  __syncthreads();
  if (w == 0) {
    float t = (lane < 8) ? smem[lane] : 0.0f;
    #pragma unroll
    for (int o = 4; o > 0; o >>= 1) t += __shfl_xor(t, o, 32);
    if (lane == 0) smem[0] = t;
  }
  __syncthreads();
  float r = smem[0];
  __syncthreads();
  return r;
}

// ---------------- weight convert + transpose: W[K,N] f32 -> WT[N,K] bf16 ----
__global__ __launch_bounds__(256) void k_wt_bf16(const float* __restrict__ W,
                                                 unsigned short* __restrict__ WT,
                                                 int K, int N) {
  size_t idx = (size_t)blockIdx.x * 256 + threadIdx.x;
  if (idx >= (size_t)K * N) return;
  int n = (int)(idx % N);
  int k = (int)(idx / N);
  WT[(size_t)n * K + k] = f2bf(W[idx]);
}

// ---------------- layernorm -> bf16 (D_MODEL = 1024, 256 thr, 4/thr) -------
__global__ __launch_bounds__(256) void k_ln_bf16(const float* __restrict__ x,
                                                 const float* __restrict__ w,
                                                 const float* __restrict__ b,
                                                 unsigned short* __restrict__ out) {
  const size_t row = blockIdx.x;
  float v[4]; float s = 0.0f;
  #pragma unroll
  for (int j = 0; j < 4; ++j) { v[j] = x[row * D_MODEL + threadIdx.x + j * 256]; s += v[j]; }
  s = block_sum256(s);
  const float mu = s * (1.0f / D_MODEL);
  float q = 0.0f;
  #pragma unroll
  for (int j = 0; j < 4; ++j) { float d = v[j] - mu; q += d * d; }
  q = block_sum256(q);
  const float inv = rsqrtf(q * (1.0f / D_MODEL) + 1e-5f);
  #pragma unroll
  for (int j = 0; j < 4; ++j) {
    int i = threadIdx.x + j * 256;
    out[row * D_MODEL + i] = f2bf((v[j] - mu) * inv * w[i] + b[i]);
  }
}

// ---------------- tiled bf16 WMMA GEMM, async-LDS double buffered -----------
// C[M,N] = A[M,K] (bf16) * BT[N,K]^T (bf16), f32 accumulate.
// Block tile 128x128, K-step 64, 8 waves (2x4), wave tile 64x32 (4x2 WMMA).
// mode 0: outf = C
// mode 1: outf = C + res
// mode 2: outb = bf16(relu(C + bias[col]))
// mode 3: outf = C + bias[col] + res
template <int K>
__global__ __launch_bounds__(256) void k_gemm_bf16(
    const unsigned short* __restrict__ A,
    const unsigned short* __restrict__ BT,
    int N, int mode,
    const float* __restrict__ bias,
    const float* __restrict__ res,
    float* __restrict__ outf,
    unsigned short* __restrict__ outb) {
  // 2 buffers x (A tile 128x64 + B tile 128x64), row stride 72 elems (144 B)
  __shared__ alignas(16) unsigned short lds[36864];   // 72 KB
  const int AS_ELEM[2] = {0, 18432};
  const int BS_ELEM[2] = {9216, 27648};

  const int m0 = blockIdx.y * 128;
  const int n0 = blockIdx.x * 128;
  const int tid  = threadIdx.x;
  const int lane = tid & 31, wave = tid >> 5;
  const int wm = wave >> 2, wn = wave & 3;     // 2 x 4 wave grid
  const int lrow = lane & 15;
  const int kb = (lane >> 4) * 8;

  const unsigned ldsBase = (unsigned)(size_t)(&lds[0]);

  // per-thread copy chunks: 4 x 16B per tile per matrix
  const unsigned short* aP[4];
  const unsigned short* bP[4];
  int coff[4];
  #pragma unroll
  for (int it = 0; it < 4; ++it) {
    int e = (tid + it * 256) * 8;              // element in 128x64 tile
    int r = e >> 6, c = e & 63;
    aP[it] = A + (size_t)(m0 + r) * K + c;
    int gn = n0 + r; gn = gn < N ? gn : N - 1;
    bP[it] = BT + (size_t)gn * K + c;
    coff[it] = r * 72 + c;                     // element offset inside tile
  }

  auto issue = [&](int buf) {
    const unsigned aB = ldsBase + (unsigned)(AS_ELEM[buf] * 2);
    const unsigned bB = ldsBase + (unsigned)(BS_ELEM[buf] * 2);
    #pragma unroll
    for (int it = 0; it < 4; ++it) {
      async_copy16(aB + (unsigned)(coff[it] * 2), aP[it]); aP[it] += 64;
      async_copy16(bB + (unsigned)(coff[it] * 2), bP[it]); bP[it] += 64;
    }
  };

  const v8f zero8 = {0.f, 0.f, 0.f, 0.f, 0.f, 0.f, 0.f, 0.f};
  v8f acc[4][2];
  #pragma unroll
  for (int i = 0; i < 4; ++i)
    #pragma unroll
    for (int j = 0; j < 2; ++j) acc[i][j] = zero8;

  // prologue: fill buffer 0
  issue(0);
  wait_async0();
  __syncthreads();

  constexpr int NIT = K / 64;
  for (int i = 0; i < NIT; ++i) {
    const int cur = i & 1;
    if (i + 1 < NIT) issue(cur ^ 1);           // overlap copy of next tile

    const int aBase = AS_ELEM[cur];
    const int bBase = BS_ELEM[cur];
    #pragma unroll
    for (int ks = 0; ks < 64; ks += 32) {
      Frag af[4], bfr[2];
      #pragma unroll
      for (int mt = 0; mt < 4; ++mt) {
        int r = wm * 64 + mt * 16 + lrow;
        af[mt].u[0] = *(const v4u*)(&lds[aBase + r * 72 + ks + kb]);
        af[mt].u[1] = *(const v4u*)(&lds[aBase + r * 72 + ks + kb + 16]);
      }
      #pragma unroll
      for (int nt = 0; nt < 2; ++nt) {
        int r = wn * 32 + nt * 16 + lrow;
        bfr[nt].u[0] = *(const v4u*)(&lds[bBase + r * 72 + ks + kb]);
        bfr[nt].u[1] = *(const v4u*)(&lds[bBase + r * 72 + ks + kb + 16]);
      }
      #pragma unroll
      for (int mt = 0; mt < 4; ++mt)
        #pragma unroll
        for (int nt = 0; nt < 2; ++nt)
          acc[mt][nt] = __builtin_amdgcn_wmma_f32_16x16x32_bf16(
              false, af[mt].v, false, bfr[nt].v, (short)0, acc[mt][nt],
              false, false);
    }

    wait_async0();                              // next-buffer writes landed
    __syncthreads();                            // all waves done reading cur
  }

  // epilogue: C layout -> lane(0..15)=col, VGPR r -> row = r + 8*(lane>=16)
  #pragma unroll
  for (int mt = 0; mt < 4; ++mt) {
    #pragma unroll
    for (int nt = 0; nt < 2; ++nt) {
      const int col = n0 + wn * 32 + nt * 16 + lrow;
      if (col < N) {
        #pragma unroll
        for (int r = 0; r < 8; ++r) {
          const int row = m0 + wm * 64 + mt * 16 + (lane >> 4) * 8 + r;
          const size_t idx = (size_t)row * N + col;
          const float v = acc[mt][nt][r];
          if (mode == 0)      outf[idx] = v;
          else if (mode == 1) outf[idx] = v + res[idx];
          else if (mode == 2) { float t = v + bias[col]; t = t > 0.f ? t : 0.f; outb[idx] = f2bf(t); }
          else                outf[idx] = v + bias[col] + res[idx];
        }
      }
    }
  }
}

// ---------------- causal conv1d (K=4) + SiLU --------------------------------
__global__ __launch_bounds__(256) void k_conv(const float* __restrict__ zx,
                                              const float* __restrict__ w,
                                              const float* __restrict__ bias,
                                              float* __restrict__ out) {
  size_t idx = (size_t)blockIdx.x * 256 + threadIdx.x;
  if (idx >= (size_t)NROWS * CONV_DIM) return;
  const int c = (int)(idx % CONV_DIM);
  const size_t row = idx / CONV_DIM;
  const int t = (int)(row % SEQLEN);
  float acc = bias[c];
  #pragma unroll
  for (int k = 0; k < 4; ++k) {
    int tt = t - 3 + k;
    if (tt >= 0) {
      size_t r2 = row - (size_t)(3 - k);
      acc += zx[r2 * D_IN_PROJ + D_INNER + c] * w[c * 4 + k];
    }
  }
  acc = acc / (1.0f + expf(-acc));  // silu
  out[idx] = acc;
}

// ---------------- SSD scan: one block per (batch, head) ---------------------
__global__ __launch_bounds__(256) void k_scan(const float* __restrict__ zx,
                                              const float* __restrict__ conv,
                                              const float* __restrict__ dt_bias,
                                              const float* __restrict__ A_log,
                                              float* __restrict__ y) {
  const int b = blockIdx.x >> 5;
  const int h = blockIdx.x & 31;
  const int tid = threadIdx.x;
  const int p  = tid >> 2;           // 0..63 head-dim
  const int nq = tid & 3;            // quarter of the n-dimension
  const int n0 = nq * 32;
  __shared__ float sB[128], sC[128], sX[64];
  __shared__ float sdA, sdt;
  const float Ah  = -expf(A_log[h]);
  const float dtb = dt_bias[h];
  float hst[32];
  #pragma unroll
  for (int i = 0; i < 32; ++i) hst[i] = 0.0f;

  for (int t = 0; t < SEQLEN; ++t) {
    const size_t row = (size_t)b * SEQLEN + t;
    const float* cr = conv + row * CONV_DIM;
    if (tid < 128) sB[tid] = cr[D_INNER + tid];
    else           sC[tid - 128] = cr[D_INNER + D_STATE + (tid - 128)];
    if (tid < 64)  sX[tid] = cr[h * HEADDIM + tid];
    if (tid == 0) {
      float raw = zx[row * D_IN_PROJ + (D_INNER + CONV_DIM) + h] + dtb;
      float dt  = raw > 20.0f ? raw : log1pf(expf(raw));   // softplus
      sdt = dt;
      sdA = expf(dt * Ah);
    }
    __syncthreads();
    const float dA  = sdA;
    const float dbx = sdt * sX[p];
    float yp = 0.0f;
    #pragma unroll
    for (int i = 0; i < 32; ++i) {
      hst[i] = hst[i] * dA + dbx * sB[n0 + i];
      yp += hst[i] * sC[n0 + i];
    }
    yp += __shfl_xor(yp, 1, 32);
    yp += __shfl_xor(yp, 2, 32);
    if (nq == 0) y[row * D_INNER + h * HEADDIM + p] = yp;
    __syncthreads();
  }
}

// ---------------- D-skip + gate(silu(z)) + RMSNorm -> bf16 ------------------
__global__ __launch_bounds__(256) void k_gate(const float* __restrict__ y,
                                              const float* __restrict__ conv,
                                              const float* __restrict__ zx,
                                              const float* __restrict__ Dskip,
                                              const float* __restrict__ rms_w,
                                              unsigned short* __restrict__ outb) {
  const size_t row = blockIdx.x;
  float g[8]; float ss = 0.0f;
  #pragma unroll
  for (int j = 0; j < 8; ++j) {
    int i = threadIdx.x + j * 256;
    float yv = y[row * D_INNER + i];
    float xv = conv[row * CONV_DIM + i];
    float zv = zx[row * D_IN_PROJ + i];
    float v  = yv + Dskip[i >> 6] * xv;
    v *= zv / (1.0f + expf(-zv));    // * silu(z)
    g[j] = v;
    ss += v * v;
  }
  ss = block_sum256(ss);
  const float scale = rsqrtf(ss * (1.0f / D_INNER) + 1e-5f);
  #pragma unroll
  for (int j = 0; j < 8; ++j) {
    int i = threadIdx.x + j * 256;
    outb[row * D_INNER + i] = f2bf(g[j] * scale * rms_w[i]);
  }
}

// ---------------------------------------------------------------------------
extern "C" void kernel_launch(void* const* d_in, const int* in_sizes, int n_in,
                              void* d_out, int out_size, void* d_ws, size_t ws_size,
                              hipStream_t stream) {
  (void)in_sizes; (void)n_in; (void)out_size; (void)ws_size;
  const float* x          = (const float*)d_in[0];
  const float* ln_w       = (const float*)d_in[1];
  const float* ln_b       = (const float*)d_in[2];
  const float* in_proj_w  = (const float*)d_in[3];
  const float* conv_w     = (const float*)d_in[4];
  const float* conv_b     = (const float*)d_in[5];
  const float* dt_bias    = (const float*)d_in[6];
  const float* A_log      = (const float*)d_in[7];
  const float* D_skip     = (const float*)d_in[8];
  const float* rms_w      = (const float*)d_in[9];
  const float* out_proj_w = (const float*)d_in[10];
  const float* mlp_w1     = (const float*)d_in[11];
  const float* mlp_b1     = (const float*)d_in[12];
  const float* mlp_w2     = (const float*)d_in[13];
  const float* mlp_b2     = (const float*)d_in[14];
  float* out = (float*)d_out;

  char* base = (char*)d_ws;
  size_t off = 0;
  auto take = [&](size_t bytes) -> void* {
    void* p = base + off;
    off += (bytes + 255) & ~(size_t)255;
    return p;
  };
  unsigned short* wtin   = (unsigned short*)take((size_t)D_IN_PROJ * D_MODEL * 2);
  unsigned short* wtout  = (unsigned short*)take((size_t)D_MODEL * D_INNER * 2);
  unsigned short* wtm1   = (unsigned short*)take((size_t)D_FF * D_MODEL * 2);
  unsigned short* wtm2   = (unsigned short*)take((size_t)D_MODEL * D_FF * 2);
  unsigned short* xnbf   = (unsigned short*)take((size_t)NROWS * D_MODEL * 2);
  float*          zx     = (float*)take((size_t)NROWS * D_IN_PROJ * 4);
  float*          convo  = (float*)take((size_t)NROWS * CONV_DIM * 4);
  float*          yscan  = (float*)take((size_t)NROWS * D_INNER * 4);
  unsigned short* ybf    = (unsigned short*)take((size_t)NROWS * D_INNER * 2);
  float*          outbuf = (float*)take((size_t)NROWS * D_MODEL * 4);
  unsigned short* h1bf   = (unsigned short*)take((size_t)NROWS * D_MODEL * 2);
  unsigned short* relubf = (unsigned short*)take((size_t)NROWS * D_FF * 2);

  // weight convert + transpose to bf16 [N,K]
  k_wt_bf16<<<((size_t)D_MODEL * D_IN_PROJ + 255) / 256, 256, 0, stream>>>(in_proj_w,  wtin, D_MODEL, D_IN_PROJ);
  k_wt_bf16<<<((size_t)D_INNER * D_MODEL + 255) / 256, 256, 0, stream>>>(out_proj_w, wtout, D_INNER, D_MODEL);
  k_wt_bf16<<<((size_t)D_MODEL * D_FF    + 255) / 256, 256, 0, stream>>>(mlp_w1,     wtm1, D_MODEL, D_FF);
  k_wt_bf16<<<((size_t)D_FF    * D_MODEL + 255) / 256, 256, 0, stream>>>(mlp_w2,     wtm2, D_FF, D_MODEL);

  // LN1 -> bf16
  k_ln_bf16<<<NROWS, 256, 0, stream>>>(x, ln_w, ln_b, xnbf);

  // in_proj: [4096,1024] x [1024,4384] -> zx (fp32)
  k_gemm_bf16<D_MODEL><<<dim3((D_IN_PROJ + 127) / 128, NROWS / 128), 256, 0, stream>>>(
      xnbf, wtin, D_IN_PROJ, 0, nullptr, nullptr, zx, nullptr);

  // conv1d + silu
  k_conv<<<((size_t)NROWS * CONV_DIM + 255) / 256, 256, 0, stream>>>(zx, conv_w, conv_b, convo);

  // SSD scan: one block per (b, h)
  k_scan<<<BATCH * N_HEADS, 256, 0, stream>>>(zx, convo, dt_bias, A_log, yscan);

  // D-skip + gate + RMSNorm -> bf16
  k_gate<<<NROWS, 256, 0, stream>>>(yscan, convo, zx, D_skip, rms_w, ybf);

  // out_proj + residual(x) -> outbuf (fp32)
  k_gemm_bf16<D_INNER><<<dim3(D_MODEL / 128, NROWS / 128), 256, 0, stream>>>(
      ybf, wtout, D_MODEL, 1, nullptr, x, outbuf, nullptr);

  // LN2 -> bf16
  k_ln_bf16<<<NROWS, 256, 0, stream>>>(outbuf, ln_w, ln_b, h1bf);

  // mlp1 + bias + relu -> bf16
  k_gemm_bf16<D_MODEL><<<dim3(D_FF / 128, NROWS / 128), 256, 0, stream>>>(
      h1bf, wtm1, D_FF, 2, mlp_b1, nullptr, nullptr, relubf);

  // mlp2 + bias + residual(outbuf) -> d_out (fp32)
  k_gemm_bf16<D_FF><<<dim3(D_MODEL / 128, NROWS / 128), 256, 0, stream>>>(
      relubf, wtm2, D_MODEL, 3, mlp_b2, outbuf, out, nullptr);
}
